// GCNNet_8340826488980
// MI455X (gfx1250) — matrix operand order repaired and checked by reading, not compiled, not run on hardware
//
#include <hip/hip_runtime.h>
#include <hip/hip_bf16.h>
#include <math.h>

#define NNODES 50000
#define NEDGES 600000
#define FIN 128
#define HDIM 128
#define NCLS 10
#define NGRAPH 128
#define BNEPS 1e-5f

typedef __attribute__((ext_vector_type(16))) _Float16 v16h;
typedef __attribute__((ext_vector_type(8)))  float    v8f;
typedef __attribute__((ext_vector_type(4)))  float    f4;

// ---------------- utility kernels ----------------

__global__ void zero_i32_kernel(int* p, int n) {
    int i = blockIdx.x * blockDim.x + threadIdx.x;
    if (i < n) p[i] = 0;
}
__global__ void zero_f32_kernel(float* p, int n) {
    int i = blockIdx.x * blockDim.x + threadIdx.x;
    if (i < n) p[i] = 0.0f;
}

// count out-degree from src list (self loops added later as +1)
__global__ void count_src_kernel(const int* __restrict__ src, int* __restrict__ deg, int e) {
    int i = blockIdx.x * blockDim.x + threadIdx.x;
    if (i < e) atomicAdd(&deg[src[i]], 1);
}

// dis[v] = rsqrt(deg[v] + 1)   (deg includes self loop => always > 0)
__global__ void compute_dis_kernel(const int* __restrict__ deg, float* __restrict__ dis, int n) {
    int i = blockIdx.x * blockDim.x + threadIdx.x;
    if (i < n) dis[i] = rsqrtf((float)(deg[i] + 1));
}

// ---------------- batchnorm (training-mode, column stats) ----------------

// partial column sums: grid = P blocks, block = 128 threads (one per column)
__global__ void colstats_partial_kernel(const float* __restrict__ in, int rows,
                                        float* __restrict__ part) {
    int j = threadIdx.x;          // column 0..127
    int p = blockIdx.x;
    int P = gridDim.x;
    float s = 0.0f, s2 = 0.0f;
    for (int r = p; r < rows; r += P) {
        float v = in[(size_t)r * 128 + j];
        s += v; s2 += v * v;
    }
    part[(size_t)p * 256 + j]       = s;
    part[(size_t)p * 256 + 128 + j] = s2;
}

// 1 block, 128 threads: fold gamma/beta into scale/shift
__global__ void colstats_finalize_kernel(const float* __restrict__ part, int P, float rows,
                                         const float* __restrict__ g, const float* __restrict__ b,
                                         float* __restrict__ scale, float* __restrict__ shift) {
    int j = threadIdx.x;
    float s = 0.0f, s2 = 0.0f;
    for (int p = 0; p < P; ++p) {
        s  += part[(size_t)p * 256 + j];
        s2 += part[(size_t)p * 256 + 128 + j];
    }
    float mean = s / rows;
    float var  = s2 / rows - mean * mean;   // biased variance, torch BN fwd
    float sc   = g[j] * rsqrtf(var + BNEPS);
    scale[j] = sc;
    shift[j] = b[j] - mean * sc;
}

// y16 = f16(x * scale[col] + shift[col])
__global__ void bn_apply_f16_kernel(const float* __restrict__ in,
                                    const float* __restrict__ scale,
                                    const float* __restrict__ shift,
                                    _Float16* __restrict__ out, int n) {
    int i = blockIdx.x * blockDim.x + threadIdx.x;
    if (i < n) {
        int c = i & 127;
        out[i] = (_Float16)(in[i] * scale[c] + shift[c]);
    }
}

// ---------------- weight conversion into WMMA B-fragment layout ----------------
// dst[((kc*8 + nt)*32 + lane)*16 + h] = w[(kc*32 + (lane>>4)*16 + h)*128 + nt*16 + (lane&15)]
__global__ void convert_bfrag_kernel(const float* __restrict__ w, _Float16* __restrict__ dst) {
    int i = blockIdx.x * blockDim.x + threadIdx.x;
    if (i >= 128 * 128) return;
    int h    = i & 15;
    int lane = (i >> 4) & 31;
    int nt   = (i >> 9) & 7;
    int kc   = i >> 12;
    int k = kc * 32 + (lane >> 4) * 16 + h;
    int n = nt * 16 + (lane & 15);
    dst[i] = (_Float16)w[k * 128 + n];
}

// ---------------- WMMA GEMM: C[M,128] = A[M,128] * B[128,128] (+bias)(+relu) ----------------
// block = 256 threads = 8 waves; wave nt owns the 16-col tile; blockIdx.x owns the 16-row tile.
// A tile (16x128 f16 = 4KB) is staged once in LDS (row-padded to 136 halves: 272B stride
// rotates 4 banks/row -> conflict-free ds_load_b128 fragment reads).
// Optionally also writes C2[idx] = dis[row]^2 * val (fused GCN self-loop init).
// M must be a multiple of 16 (50000 and 128 both are).
__global__ void gemm128_wmma_kernel(const _Float16* __restrict__ A,
                                    const _Float16* __restrict__ Bfrag,
                                    float* __restrict__ C, int M,
                                    const float* __restrict__ bias, int relu,
                                    float* __restrict__ C2, const float* __restrict__ dis) {
    __shared__ _Float16 At[16 * 136];
    int mtile = blockIdx.x;
    int t = threadIdx.x;

    // stage A tile: each of 256 threads moves 8 halves (16B)
    {
        int row = t >> 4;
        int col = (t & 15) * 8;
        f4 v = *reinterpret_cast<const f4*>(A + (size_t)(mtile * 16 + row) * 128 + col);
        *reinterpret_cast<f4*>(&At[row * 136 + col]) = v;
    }
    __syncthreads();

    int nt   = t >> 5;        // wave id 0..7 -> column tile
    int lane = t & 31;
    int r15  = lane & 15;
    int hgrp = lane >> 4;

    const _Float16* arow = &At[r15 * 136 + hgrp * 16];
    const _Float16* bptr = Bfrag + ((size_t)nt * 32 + lane) * 16;

    v8f acc = {};
    #pragma unroll
    for (int kc = 0; kc < 4; ++kc) {
        v16h a, b;
        f4* ap = reinterpret_cast<f4*>(&a);
        f4* bp = reinterpret_cast<f4*>(&b);
        ap[0] = *reinterpret_cast<const f4*>(arow + kc * 32);       // K = hgrp*16   .. +7
        ap[1] = *reinterpret_cast<const f4*>(arow + kc * 32 + 8);   // K = hgrp*16+8 .. +15
        bp[0] = *reinterpret_cast<const f4*>(bptr + (size_t)kc * 4096);
        bp[1] = *reinterpret_cast<const f4*>(bptr + (size_t)kc * 4096 + 8);
        acc = __builtin_amdgcn_wmma_f32_16x16x32_f16(false, a, false, b,
                                                     (short)0, acc, false, false);
    }

    int col = nt * 16 + r15;          // N index for this lane
    float bv = bias ? bias[col] : 0.0f;
    #pragma unroll
    for (int v = 0; v < 8; ++v) {
        int m = hgrp * 8 + v;         // M index per C/D layout
        int grow = mtile * 16 + m;
        float val = acc[v] + bv;
        if (relu) val = fmaxf(val, 0.0f);
        size_t idx = (size_t)grow * 128 + col;
        C[idx] = val;
        if (C2) {                     // fused self-loop init: dis[v]^2 * z
            float dd = dis[grow];
            C2[idx] = dd * dd * val;
        }
    }
}

// ---------------- GCN propagate ----------------

// one wave per edge (grid-stride): 4 floats per lane, coalesced 512B per edge
__global__ void edge_scatter_kernel(const int* __restrict__ src, const int* __restrict__ dst,
                                    const float* __restrict__ dis,
                                    const float* __restrict__ z, float* __restrict__ out, int e) {
    int wid  = (blockIdx.x * blockDim.x + threadIdx.x) >> 5;
    int lane = threadIdx.x & 31;
    int nw   = (gridDim.x * blockDim.x) >> 5;
    for (int ed = wid; ed < e; ed += nw) {
        int s = src[ed], d = dst[ed];
        float nrm = dis[s] * dis[d];
        f4 m = *reinterpret_cast<const f4*>(z + (size_t)s * 128 + lane * 4);
        float* op = out + (size_t)d * 128 + lane * 4;
        unsafeAtomicAdd(op + 0, nrm * m[0]);
        unsafeAtomicAdd(op + 1, nrm * m[1]);
        unsafeAtomicAdd(op + 2, nrm * m[2]);
        unsafeAtomicAdd(op + 3, nrm * m[3]);
    }
}

// h = relu(h + bias[col]) in place
__global__ void bias_relu_kernel(float* __restrict__ h, const float* __restrict__ bias, int n) {
    int i = blockIdx.x * blockDim.x + threadIdx.x;
    if (i < n) h[i] = fmaxf(h[i] + bias[i & 127], 0.0f);
}

// global_add_pool: hg[batch[v]] += h[v]
__global__ void pool_kernel(const float* __restrict__ h, const int* __restrict__ batch,
                            float* __restrict__ hg, int n) {
    int wid  = (blockIdx.x * blockDim.x + threadIdx.x) >> 5;
    int lane = threadIdx.x & 31;
    int nw   = (gridDim.x * blockDim.x) >> 5;
    for (int v = wid; v < n; v += nw) {
        int g = batch[v];
        f4 m = *reinterpret_cast<const f4*>(h + (size_t)v * 128 + lane * 4);
        float* op = hg + (size_t)g * 128 + lane * 4;
        unsafeAtomicAdd(op + 0, m[0]);
        unsafeAtomicAdd(op + 1, m[1]);
        unsafeAtomicAdd(op + 2, m[2]);
        unsafeAtomicAdd(op + 3, m[3]);
    }
}

// ---------------- classifier head: BN(inline) @ w_cls + b_cls -> log_softmax ----------------
__global__ void head_kernel(const float* __restrict__ hg2,
                            const float* __restrict__ scale, const float* __restrict__ shift,
                            const float* __restrict__ wcls, const float* __restrict__ bcls,
                            float* __restrict__ out) {
    __shared__ float v[128];
    __shared__ float logits[NCLS];
    int g = blockIdx.x, t = threadIdx.x;
    v[t] = scale[t] * hg2[(size_t)g * 128 + t] + shift[t];
    __syncthreads();
    if (t < NCLS) {
        float s = bcls[t];
        for (int k = 0; k < 128; ++k) s += v[k] * wcls[k * NCLS + t];
        logits[t] = s;
    }
    __syncthreads();
    if (t == 0) {
        float mx = logits[0];
        for (int c = 1; c < NCLS; ++c) mx = fmaxf(mx, logits[c]);
        float se = 0.0f;
        for (int c = 0; c < NCLS; ++c) se += __expf(logits[c] - mx);
        float l = logf(se);
        for (int c = 0; c < NCLS; ++c) out[g * NCLS + c] = logits[c] - mx - l;
    }
}

// ---------------- host orchestration ----------------

extern "C" void kernel_launch(void* const* d_in, const int* in_sizes, int n_in,
                              void* d_out, int out_size, void* d_ws, size_t ws_size,
                              hipStream_t stream) {
    (void)in_sizes; (void)n_in; (void)out_size; (void)ws_size;

    const float* x        = (const float*)d_in[0];
    const int*   ei       = (const int*)d_in[1];
    const int*   src      = ei;
    const int*   dst      = ei + NEDGES;
    const int*   batch    = (const int*)d_in[2];
    const float* bn_feat_g = (const float*)d_in[3];
    const float* bn_feat_b = (const float*)d_in[4];
    const float* w_feat    = (const float*)d_in[5];
    const float* bn_conv_g = (const float*)d_in[6];
    const float* bn_conv_b = (const float*)d_in[7];
    const float* w_conv    = (const float*)d_in[8];
    const float* b_conv    = (const float*)d_in[9];
    const float* bn_fc_g   = (const float*)d_in[10];
    const float* bn_fc_b   = (const float*)d_in[11];
    const float* w_fc      = (const float*)d_in[12];
    const float* b_fc      = (const float*)d_in[13];
    const float* bn_hid_g  = (const float*)d_in[14];
    const float* bn_hid_b  = (const float*)d_in[15];
    const float* w_cls     = (const float*)d_in[16];
    const float* b_cls     = (const float*)d_in[17];
    float* out = (float*)d_out;

    // workspace bump allocator (256B aligned)
    char* wsb = (char*)d_ws;
    size_t off = 0;
    auto alloc = [&](size_t bytes) -> void* {
        void* p = wsb + off;
        off = (off + bytes + 255) & ~(size_t)255;
        return p;
    };

    int*      deg   = (int*)alloc((size_t)NNODES * 4);
    float*    dis   = (float*)alloc((size_t)NNODES * 4);
    float*    part  = (float*)alloc((size_t)512 * 256 * 4);
    float*    scale = (float*)alloc(128 * 4);
    float*    shift = (float*)alloc(128 * 4);
    _Float16* a16   = (_Float16*)alloc((size_t)NNODES * 128 * 2);
    float*    hA    = (float*)alloc((size_t)NNODES * 128 * 4);
    float*    zb    = (float*)alloc((size_t)NNODES * 128 * 4);
    _Float16* wfrag = (_Float16*)alloc((size_t)5 * 128 * 128 * 2);
    float*    hg    = (float*)alloc((size_t)NGRAPH * 128 * 4);
    _Float16* hg16  = (_Float16*)alloc((size_t)NGRAPH * 128 * 2);
    float*    hg2   = (float*)alloc((size_t)NGRAPH * 128 * 4);

    const int NE = NNODES * 128;               // 6.4M elements
    dim3 b256(256);
    int gNE   = (NE + 255) / 256;
    int gNode = (NNODES + 255) / 256;
    int gEdge = (NEDGES + 255) / 256;
    const int STATS_P = 512;

    // --- degree / normalization coefficients ---
    zero_i32_kernel<<<gNode, b256, 0, stream>>>(deg, NNODES);
    count_src_kernel<<<gEdge, b256, 0, stream>>>(src, deg, NEDGES);
    compute_dis_kernel<<<gNode, b256, 0, stream>>>(deg, dis, NNODES);

    // --- convert all big weights into WMMA B-fragment f16 layout ---
    int gW = (128 * 128 + 255) / 256;
    convert_bfrag_kernel<<<gW, b256, 0, stream>>>(w_feat, wfrag + 0 * 16384);
    for (int i = 0; i < 3; ++i)
        convert_bfrag_kernel<<<gW, b256, 0, stream>>>(w_conv + (size_t)i * 16384,
                                                      wfrag + (size_t)(1 + i) * 16384);
    convert_bfrag_kernel<<<gW, b256, 0, stream>>>(w_fc, wfrag + 4 * 16384);

    // --- feature BN + linear + relu ---
    colstats_partial_kernel<<<STATS_P, 128, 0, stream>>>(x, NNODES, part);
    colstats_finalize_kernel<<<1, 128, 0, stream>>>(part, STATS_P, (float)NNODES,
                                                    bn_feat_g, bn_feat_b, scale, shift);
    bn_apply_f16_kernel<<<gNE, b256, 0, stream>>>(x, scale, shift, a16, NE);
    gemm128_wmma_kernel<<<NNODES / 16, b256, 0, stream>>>(a16, wfrag, hA, NNODES,
                                                          nullptr, 1, nullptr, nullptr);

    // --- 3 GCN layers ---
    for (int i = 0; i < 3; ++i) {
        colstats_partial_kernel<<<STATS_P, 128, 0, stream>>>(hA, NNODES, part);
        colstats_finalize_kernel<<<1, 128, 0, stream>>>(part, STATS_P, (float)NNODES,
                                                        bn_conv_g + i * 128, bn_conv_b + i * 128,
                                                        scale, shift);
        bn_apply_f16_kernel<<<gNE, b256, 0, stream>>>(hA, scale, shift, a16, NE);
        // z = BN(h) @ w_conv[i]; fused: hA = dis^2 * z (self-loop term)
        gemm128_wmma_kernel<<<NNODES / 16, b256, 0, stream>>>(a16, wfrag + (size_t)(1 + i) * 16384,
                                                              zb, NNODES, nullptr, 0, hA, dis);
        edge_scatter_kernel<<<2048, b256, 0, stream>>>(src, dst, dis, zb, hA, NEDGES);
        bias_relu_kernel<<<gNE, b256, 0, stream>>>(hA, b_conv + i * 128, NE);
    }

    // --- global add pool ---
    int gHg = (NGRAPH * 128 + 255) / 256;
    zero_f32_kernel<<<gHg, b256, 0, stream>>>(hg, NGRAPH * 128);
    pool_kernel<<<1024, b256, 0, stream>>>(hA, batch, hg, NNODES);

    // --- FC layer: relu(BN(hg) @ w_fc + b_fc) ---
    colstats_partial_kernel<<<8, 128, 0, stream>>>(hg, NGRAPH, part);
    colstats_finalize_kernel<<<1, 128, 0, stream>>>(part, 8, (float)NGRAPH,
                                                    bn_fc_g, bn_fc_b, scale, shift);
    bn_apply_f16_kernel<<<gHg, b256, 0, stream>>>(hg, scale, shift, hg16, NGRAPH * 128);
    gemm128_wmma_kernel<<<NGRAPH / 16, b256, 0, stream>>>(hg16, wfrag + 4 * 16384,
                                                          hg2, NGRAPH, b_fc, 1,
                                                          nullptr, nullptr);

    // --- head: BN(hg2) @ w_cls + b_cls -> log_softmax ---
    colstats_partial_kernel<<<8, 128, 0, stream>>>(hg2, NGRAPH, part);
    colstats_finalize_kernel<<<1, 128, 0, stream>>>(part, 8, (float)NGRAPH,
                                                    bn_hid_g, bn_hid_b, scale, shift);
    head_kernel<<<NGRAPH, 128, 0, stream>>>(hg2, scale, shift, w_cls, b_cls, out);
}
